// OLS_loss_87540023427607
// MI455X (gfx1250) — matrix operand order repaired and checked by reading, not compiled, not run on hardware
//
#include <hip/hip_runtime.h>

typedef __attribute__((ext_vector_type(16))) _Float16 v16h;
typedef __attribute__((ext_vector_type(8)))  float    v8f;

#define OLS_C 1000   // number of classes (loss_lams is C x C)

__global__ void ols_zero_kernel(float* __restrict__ p, int n) {
  int i = blockIdx.x * blockDim.x + threadIdx.x;
  int stride = gridDim.x * blockDim.x;
  for (; i < n; i += stride) p[i] = 0.0f;
}

// One wave32 per 16-row tile; 8 waves (256 threads) per block.
__launch_bounds__(256)
__global__ void ols_loss_kernel(const float* __restrict__ outp,
                                const int*   __restrict__ target,
                                const float* __restrict__ lams,
                                float* __restrict__ loss,      // d_out[0]
                                float* __restrict__ lams_up,   // d_out + 1        [C*C]
                                float* __restrict__ cnt_up,    // d_out + 1 + C*C  [C]
                                int Btot) {
  constexpr int C = OLS_C;
  constexpr int NFULL = C / 32;                   // 31 full K=32 chunks
  const int lane = threadIdx.x & 31;
  const int wave = threadIdx.x >> 5;

  __shared__ float s_logZ[8][16];
  __shared__ int   s_t[8][16];
  __shared__ int   s_amax[8][16];

  const int tile = blockIdx.x * 8 + wave;
  const int row0 = tile * 16;
  const float invB = 1.0f / (float)Btot;

  // ---- stage targets for this tile ----
  if (lane < 16) {
    int r = row0 + lane;
    s_t[wave][lane] = (r < Btot) ? target[r] : 0;
  }

  // ---- phase 1: per-row online softmax (max, first-argmax, sumexp) ----
  for (int m = 0; m < 16; ++m) {
    int r = row0 + m;
    float vmax = -3.402823466e38f;
    float ssum = 0.0f;
    int   vidx = 0x7fffffff;
    if (r < Btot) {
      const float* xr = outp + (size_t)r * C;
      for (int j = lane; j < C; j += 32) {
        if (m < 15) __builtin_prefetch(xr + C + j, 0, 1);  // -> global_prefetch_b8
        float x = xr[j];
        if (x > vmax) {
          ssum = ssum * __expf(vmax - x) + 1.0f;
          vmax = x; vidx = j;
        } else {
          ssum += __expf(x - vmax);
        }
      }
    }
    // wave32 butterfly merge of (max, first-index, scaled-sum)
    for (int off = 16; off > 0; off >>= 1) {
      float om = __shfl_xor(vmax, off, 32);
      float os = __shfl_xor(ssum, off, 32);
      int   oi = __shfl_xor(vidx, off, 32);
      if (om > vmax) {
        ssum = ssum * __expf(vmax - om) + os;
        vmax = om; vidx = oi;
      } else if (om == vmax) {
        ssum += os;
        vidx = (oi < vidx) ? oi : vidx;
      } else {
        ssum += os * __expf(om - vmax);
      }
    }
    if (lane == 0) {
      s_logZ[wave][m] = vmax + __logf(ssum);
      s_amax[wave][m] = vidx;
    }
  }
  __syncthreads();

  // ---- phase 2: soft-CE dots via v_wmma_f32_16x16x32_f16 ----
  // A (16x32 f16), lane layout per ISA 7.12.2:
  //   M = lane&15 ; element e=0..7  -> K = kloA + e       (kloA = 0|8)
  //                 element e=8..15 -> K = 16 + kloA + e-8
  // B (32x16 f16): N = lane&15 ; same K layout as A (kloB = 0|8)
  const int   mA     = lane & 15;
  const int   kloA   = (lane >> 4) << 3;          // 0 or 8
  const int   rowA   = row0 + mA;
  const bool  rowAok = (rowA < Btot);
  const float lzA    = s_logZ[wave][mA];
  const float* xA    = outp + (size_t)(rowAok ? rowA : 0) * C;

  const int   nB     = lane & 15;
  const int   tB     = s_t[wave][nB];
  const float* yB    = lams + (size_t)tB * C;
  const int   kloB   = (lane >> 4) << 3;          // 0 or 8

  v8f acc = {};
  for (int ch = 0; ch < NFULL; ++ch) {
    const int cb = ch * 32;
    // A and B: two 8-float runs each — all 16B-aligned -> b128 loads
    float4 a0 = *(const float4*)(xA + cb + kloA);
    float4 a1 = *(const float4*)(xA + cb + kloA + 4);
    float4 a2 = *(const float4*)(xA + cb + kloA + 16);
    float4 a3 = *(const float4*)(xA + cb + kloA + 20);
    float4 b0 = *(const float4*)(yB + cb + kloB);
    float4 b1 = *(const float4*)(yB + cb + kloB + 4);
    float4 b2 = *(const float4*)(yB + cb + kloB + 16);
    float4 b3 = *(const float4*)(yB + cb + kloB + 20);
    float av[16] = {a0.x, a0.y, a0.z, a0.w, a1.x, a1.y, a1.z, a1.w,
                    a2.x, a2.y, a2.z, a2.w, a3.x, a3.y, a3.z, a3.w};
    float bv[16] = {b0.x, b0.y, b0.z, b0.w, b1.x, b1.y, b1.z, b1.w,
                    b2.x, b2.y, b2.z, b2.w, b3.x, b3.y, b3.z, b3.w};
    v16h a, b;
#pragma unroll
    for (int e = 0; e < 16; ++e) {
      a[e] = (_Float16)(av[e] - lzA);
      b[e] = (_Float16)bv[e];
    }
    acc = __builtin_amdgcn_wmma_f32_16x16x32_f16(false, a, false, b,
                                                 (short)0, acc, false, false);
  }
  { // tail chunk (cols 992..999), branchless: clamp address, select zero
    const int cb = NFULL * 32;
    v16h a, b;
#pragma unroll
    for (int e = 0; e < 16; ++e) {
      int kA  = (e < 8) ? (kloA + e) : (16 + kloA + (e - 8));
      int cA  = cb + kA;
      int cAc = (cA < C) ? cA : (C - 1);
      float xv = xA[cAc] - lzA;
      a[e] = (_Float16)((rowAok && cA < C) ? xv : 0.0f);
      int cB  = cb + ((e < 8) ? (kloB + e) : (16 + kloB + (e - 8)));
      int cBc = (cB < C) ? cB : (C - 1);
      float yv = yB[cBc];
      b[e] = (_Float16)((cB < C) ? yv : 0.0f);
    }
    acc = __builtin_amdgcn_wmma_f32_16x16x32_f16(false, a, false, b,
                                                 (short)0, acc, false, false);
  }

  // Diagonal D[m][m] = dot(logp_m, lams[t_m]):
  //   m in 0..7  -> lane m     , acc[m]
  //   m in 8..15 -> lane m+16  , acc[m-8]
  const bool hasDiag = (lane < 8) || (lane >= 24);
  const int  dm = (lane < 8) ? lane : (lane - 16);
  const int  dr = (lane < 8) ? lane : (lane - 24);
  float dotm = acc[0];
  if (dr == 1) dotm = acc[1];
  if (dr == 2) dotm = acc[2];
  if (dr == 3) dotm = acc[3];
  if (dr == 4) dotm = acc[4];
  if (dr == 5) dotm = acc[5];
  if (dr == 6) dotm = acc[6];
  if (dr == 7) dotm = acc[7];

  float contrib = 0.0f;
  if (hasDiag) {
    int r = row0 + dm;
    if (r < Btot) {
      int   t  = s_t[wave][dm];
      float lz = s_logZ[wave][dm];
      float logpt = outp[(size_t)r * C + t] - lz;     // hard-CE gather
      contrib = -0.5f * invB * (logpt + dotm);        // 0.5*CE + 0.5*softCE
    }
  }
  for (int off = 16; off > 0; off >>= 1)
    contrib += __shfl_xor(contrib, off, 32);
  if (lane == 0) atomicAdd(loss, contrib);

  // ---- phase 3: scatter prob vectors of correctly-predicted rows ----
  for (int m = 0; m < 16; ++m) {
    int r = row0 + m;
    if (r >= Btot) break;
    int t = s_t[wave][m];
    if (s_amax[wave][m] == t) {
      float lz = s_logZ[wave][m];
      const float* xr  = outp + (size_t)r * C;
      float*       dst = lams_up + (size_t)t * C;
      for (int j = lane; j < C; j += 32)
        atomicAdd(dst + j, __expf(xr[j] - lz));       // prob = exp(logp)
      if (lane == 0) atomicAdd(cnt_up + t, 1.0f);
    }
  }
}

extern "C" void kernel_launch(void* const* d_in, const int* in_sizes, int n_in,
                              void* d_out, int out_size, void* d_ws, size_t ws_size,
                              hipStream_t stream) {
  (void)n_in; (void)d_ws; (void)ws_size;
  const float* outp   = (const float*)d_in[0];
  const int*   target = (const int*)d_in[1];
  const float* lams   = (const float*)d_in[2];
  float* out = (float*)d_out;

  const int Btot = in_sizes[1];                 // 32768
  // d_out = [loss(1)] [lams_update(C*C)] [cnt_update(C)] — zero each call
  int zb = (out_size + 255) / 256;
  if (zb > 1024) zb = 1024;
  ols_zero_kernel<<<zb, 256, 0, stream>>>(out, out_size);

  int tiles  = (Btot + 15) / 16;                // 16 rows per wave
  int blocks = (tiles + 7) / 8;                 // 8 waves per block
  ols_loss_kernel<<<blocks, 256, 0, stream>>>(
      outp, target, lams,
      out, out + 1, out + 1 + (size_t)OLS_C * OLS_C, Btot);
}
